// StudentNet_10548439679215
// MI455X (gfx1250) — matrix operand order repaired and checked by reading, not compiled, not run on hardware
//
#include <hip/hip_runtime.h>
#include <hip/hip_bf16.h>
#include <math.h>

// ---------------------------------------------------------------------------
// MI455X (gfx1250) wave32 WMMA implementation of the reference transformer.
// All dense math runs on v_wmma_f32_16x16x32_f16 (f16 in, f32 accumulate).
// Attention is flash-style (online softmax), logits never touch HBM.
// Global->LDS staging uses GLOBAL_LOAD_ASYNC_TO_LDS_B128 (ASYNCcnt) when the
// toolchain exposes the builtins; otherwise falls back to synchronous loads.
// ---------------------------------------------------------------------------

typedef _Float16 half_t;
typedef __attribute__((ext_vector_type(16))) _Float16 v16h;
typedef __attribute__((ext_vector_type(8)))  float    v8f;
typedef __attribute__((ext_vector_type(4)))  int      v4i;

constexpr int Bc = 32, Nc = 8, Pc = 6, Dc = 96;
constexpr int Ec = 128, Hc = 8, Ic = 64, Kc = 32, NLc = 2;
constexpr int Lc = Nc * Dc;          // 768
constexpr int Rows = Bc * Lc;        // 24576 sequence rows
constexpr int OUTc = 128;
constexpr int KF = Lc * Ec;          // 98304 final K
constexpr int KSPLIT = 48;           // split-K for the tiny-M final GEMM

#if __has_builtin(__builtin_amdgcn_global_load_async_to_lds_b128) && \
    __has_builtin(__builtin_amdgcn_s_wait_asynccnt)
#define HAVE_ASYNC_LDS 1
#else
#define HAVE_ASYNC_LDS 0
#endif

#if HAVE_ASYNC_LDS
typedef __attribute__((address_space(1))) v4i gbl_v4i;
typedef __attribute__((address_space(3))) v4i lds_v4i;
__device__ __forceinline__ void async_b128(void* lds_dst, const void* gsrc) {
  __builtin_amdgcn_global_load_async_to_lds_b128(
      (gbl_v4i*)gsrc, (lds_v4i*)lds_dst, 0, 0);
}
#endif

__device__ __forceinline__ int kbase_a(int v) { return 2 * (v & 3) + 16 * (v >> 2); }

// ---------------- prototype row-normalize: pn[d,k,p] -----------------------
__global__ __launch_bounds__(128) void proto_norm_kernel(
    const float* __restrict__ proto, float* __restrict__ pn) {
  int i = blockIdx.x * 128 + threadIdx.x;
  if (i >= Dc * Kc) return;
  const float* s = proto + i * Pc;
  float n2 = 0.f;
  for (int p = 0; p < Pc; ++p) n2 += s[p] * s[p];
  float inv = 1.f / fmaxf(sqrtf(n2), 1e-12f);
  for (int p = 0; p < Pc; ++p) pn[i * Pc + p] = s[p] * inv;
}

// ------- fused: x-norm, cosine argmax, embed, proj, pos-encoding -----------
__global__ __launch_bounds__(128) void embed_kernel(
    const float* __restrict__ x, const float* __restrict__ pn,
    const float* __restrict__ emb_w, const float* __restrict__ emb_b,
    const float* __restrict__ proj_w, const float* __restrict__ proj_b,
    float* __restrict__ seqF, half_t* __restrict__ seqH) {
  __shared__ float xv[Pc];
  __shared__ float xn[Pc];
  __shared__ float simsh[Kc];
  __shared__ int bks;
  const int r = blockIdx.x;       // row of x_flat [24576, 6]
  const int t = threadIdx.x;      // embedding channel e
  const int d = r % Dc;
  if (t < Pc) xv[t] = x[r * Pc + t];
  __syncthreads();
  if (t == 0) {
    float n2 = 0.f;
    for (int p = 0; p < Pc; ++p) n2 += xv[p] * xv[p];
    float inv = 1.f / fmaxf(sqrtf(n2), 1e-12f);
    for (int p = 0; p < Pc; ++p) xn[p] = xv[p] * inv;
  }
  __syncthreads();
  if (t < Kc) {
    const float* pr = pn + (d * Kc + t) * Pc;
    float s = 0.f;
    for (int p = 0; p < Pc; ++p) s += xn[p] * pr[p];
    simsh[t] = s;
  }
  __syncthreads();
  if (t == 0) {
    float best = simsh[0]; int bi = 0;
    for (int k = 1; k < Kc; ++k) if (simsh[k] > best) { best = simsh[k]; bi = k; }
    bks = bi;
  }
  __syncthreads();
  const float* er  = emb_w + t * Pc;
  const float* prj = proj_w + t * Pc;
  const float* sel = pn + (d * Kc + bks) * Pc;
  float o = emb_b[t] + proj_b[t];
  for (int p = 0; p < Pc; ++p) o += xv[p] * er[p] + sel[p] * prj[p];
  const int l = r % Lc;
  float ang = (float)l * __expf(-(float)(t & ~1) * (logf(10000.f) / (float)Ec));
  o += (t & 1) ? cosf(ang) : sinf(ang);
  const size_t idx = (size_t)r * Ec + t;
  seqF[idx] = o;
  seqH[idx] = (half_t)o;
}

// ---------------- generic WMMA GEMM:  C = A[M,K](f16) * W[N,K](f32)^T ------
// EPI: 0 = +bias -> f16 out   (QKV)
//      1 = +bias -> f32 out   (out_proj / ffn2 residual input)
//      2 = +bias, ELU -> f16  (ffn1)
//      3 = split-K partial -> f32 partial[z][M][N]  (final head)
template <int EPI>
__global__ __launch_bounds__(128) void gemm_wmma_kernel(
    const half_t* __restrict__ A, const float* __restrict__ W,
    const float* __restrict__ bias, float* __restrict__ outF,
    half_t* __restrict__ outH, int M, int N, int Ktot) {
  __shared__ __align__(16) half_t Ash[16 * 128];
  const int tid  = threadIdx.x;
  const int wave = tid >> 5;
  const int lane = tid & 31;
  const int mt   = blockIdx.x;
  const int nt   = blockIdx.y * 4 + wave;
  const int Klen = Ktot / (int)gridDim.z;
  const int k0   = (int)blockIdx.z * Klen;
  const int row0 = mt * 16;
  const int m    = lane & 15;
  const int g    = lane >> 4;
  const int n    = nt * 16 + m;          // output column / W row
  const bool active = (nt * 16) < N;
  v8f acc = {};
  for (int kc = 0; kc < Klen; kc += 128) {
    const int clen = (Klen - kc) < 128 ? (Klen - kc) : 128;
#if HAVE_ASYNC_LDS
    // async global->LDS tile copy, 16B per lane-op, tracked on ASYNCcnt
    for (int i = tid * 8; i < 16 * clen; i += 128 * 8) {
      const int rr = i / clen;
      const int cc = i - rr * clen;
      async_b128(&Ash[rr * 128 + cc],
                 &A[(size_t)(row0 + rr) * Ktot + (k0 + kc + cc)]);
    }
    __builtin_amdgcn_s_wait_asynccnt(0);
#else
    for (int i = tid; i < 16 * clen; i += 128) {
      const int rr = i / clen;
      const int cc = i - rr * clen;
      Ash[rr * 128 + cc] = A[(size_t)(row0 + rr) * Ktot + (k0 + kc + cc)];
    }
#endif
    __syncthreads();
    if (active) {
      const float* wrow = W + (size_t)n * Ktot + (k0 + kc);
      if (kc + 128 < Klen) __builtin_prefetch(wrow + 128, 0, 0);
      for (int ks = 0; ks < clen; ks += 32) {
        v16h af, bf;
#pragma unroll
        for (int j = 0; j < 16; ++j) {       // A frag per ISA 16-bit 16x32 table
          const int v  = j >> 1;
          const int kk = ks + kbase_a(v) + 8 * g + (j & 1);
          af[j] = Ash[m * 128 + kk];
        }
#pragma unroll
        for (int j = 0; j < 16; ++j) {       // B frag: K per ISA 16-bit table (f32 src)
          bf[j] = (half_t)wrow[ks + (j & 7) + 8 * g + 16 * (j >> 3)];
        }
        acc = __builtin_amdgcn_wmma_f32_16x16x32_f16(
            false, af, false, bf, (short)0, acc, false, false);
      }
    }
    __syncthreads();
  }
  if (!active) return;
#pragma unroll
  for (int r = 0; r < 8; ++r) {
    const int row = row0 + r + 8 * g;
    const size_t idx = (size_t)row * N + n;
    if constexpr (EPI == 0) {
      outH[idx] = (half_t)(acc[r] + bias[n]);
    } else if constexpr (EPI == 1) {
      outF[idx] = acc[r] + bias[n];
    } else if constexpr (EPI == 2) {
      float val = acc[r] + bias[n];
      val = val > 0.f ? val : (__expf(val) - 1.f);
      outH[idx] = (half_t)val;
    } else {
      outF[(size_t)blockIdx.z * M * N + idx] = acc[r];
    }
  }
}

// ---------------- flash attention: one (b,h) per block, 4 waves ------------
__global__ __launch_bounds__(128) void attn_kernel(
    const half_t* __restrict__ qkv, half_t* __restrict__ ctx) {
  __shared__ __align__(16) half_t ksh[Lc * 16];   // 24 KB
  __shared__ __align__(16) half_t vsh[Lc * 16];   // 24 KB
  __shared__ __align__(16) half_t psh[4][16][32]; // 4 KB (P tile C->A relayout)
  const int b = blockIdx.x >> 3;
  const int h = blockIdx.x & 7;
  const int tid = threadIdx.x, wave = tid >> 5, lane = tid & 31;
  const size_t base = (size_t)b * Lc * (3 * Ec);
#if HAVE_ASYNC_LDS
  // stage k,v head-slices: 32B rows strided 768B in qkv -> packed LDS
  for (int i = tid * 8; i < Lc * 16; i += 128 * 8) {
    const int l = i >> 4, d = i & 15;   // d in {0, 8}
    async_b128(&ksh[i], &qkv[base + (size_t)l * (3 * Ec) + Ec     + h * 16 + d]);
    async_b128(&vsh[i], &qkv[base + (size_t)l * (3 * Ec) + 2 * Ec + h * 16 + d]);
  }
  __builtin_amdgcn_s_wait_asynccnt(0);
#else
  for (int i = tid; i < Lc * 16; i += 128) {
    const int l = i >> 4, d = i & 15;
    ksh[i] = qkv[base + (size_t)l * (3 * Ec) + Ec     + h * 16 + d];
    vsh[i] = qkv[base + (size_t)l * (3 * Ec) + 2 * Ec + h * 16 + d];
  }
#endif
  __syncthreads();
  const int m = lane & 15, g = lane >> 4;
  for (int qt = wave; qt < Lc / 16; qt += 4) {
    v16h aq;                                 // q tile: K = head dim, pad 16..31 = 0
#pragma unroll
    for (int j = 0; j < 16; ++j) {
      const int v  = j >> 1;
      const int kk = kbase_a(v) + 8 * g + (j & 1);
      aq[j] = (kk < 16)
          ? qkv[base + (size_t)(qt * 16 + m) * (3 * Ec) + h * 16 + kk]
          : (half_t)0.f;
    }
    float mi[8], li[8];
    v8f acc = {};
#pragma unroll
    for (int r = 0; r < 8; ++r) { mi[r] = -1e30f; li[r] = 0.f; }

    for (int kt = 0; kt < Lc / 32; ++kt) {   // 32 keys per step
      const int kb0 = kt * 32;
      auto loadBk = [&](int kb) {            // k^T frag: K = head dim (pad), N = key
        v16h bf;
        const int key = kb + (lane & 15);
#pragma unroll
        for (int j = 0; j < 16; ++j) {
          const int kk = (j & 7) + 8 * g + 16 * (j >> 3);
          bf[j] = (kk < 16) ? ksh[key * 16 + kk] : (half_t)0.f;
        }
        return bf;
      };
      v8f zc = {};
      v8f s0 = __builtin_amdgcn_wmma_f32_16x16x32_f16(
          false, aq, false, loadBk(kb0),      (short)0, zc, false, false);
      v8f s1 = __builtin_amdgcn_wmma_f32_16x16x32_f16(
          false, aq, false, loadBk(kb0 + 16), (short)0, zc, false, false);
#pragma unroll
      for (int r = 0; r < 8; ++r) {
        float a0 = s0[r] * 0.25f;            // scale = 1/sqrt(16)
        float a1 = s1[r] * 0.25f;
        float mx = fmaxf(a0, a1);
        for (int off = 1; off < 16; off <<= 1) mx = fmaxf(mx, __shfl_xor(mx, off));
        const float mnew = fmaxf(mi[r], mx);
        const float corr = __expf(mi[r] - mnew);
        const float p0 = __expf(a0 - mnew);
        const float p1 = __expf(a1 - mnew);
        float rs = p0 + p1;
        for (int off = 1; off < 16; off <<= 1) rs += __shfl_xor(rs, off);
        li[r] = li[r] * corr + rs;
        acc[r] = acc[r] * corr;
        mi[r] = mnew;
        const int row = r + 8 * g;
        psh[wave][row][lane & 15]        = (half_t)p0;
        psh[wave][row][(lane & 15) + 16] = (half_t)p1;
      }
      __threadfence_block();                 // s_wait_dscnt: wave-local LDS relayout
      v16h ap, bv;
#pragma unroll
      for (int j = 0; j < 16; ++j) {         // P as A frag: K = key offset 0..31
        const int v  = j >> 1;
        const int kk = kbase_a(v) + 8 * g + (j & 1);
        ap[j] = psh[wave][m][kk];
      }
#pragma unroll
      for (int j = 0; j < 16; ++j) {         // V as B frag: K = key, N = head dim
        bv[j] = vsh[(kb0 + (j & 7) + 8 * g + 16 * (j >> 3)) * 16 + (lane & 15)];
      }
      acc = __builtin_amdgcn_wmma_f32_16x16x32_f16(
          false, ap, false, bv, (short)0, acc, false, false);
    }
#pragma unroll
    for (int r = 0; r < 8; ++r) {
      const int row = qt * 16 + r + 8 * g;
      const float o = acc[r] / li[r];
      ctx[((size_t)b * Lc + row) * Ec + h * 16 + (lane & 15)] = (half_t)o;
    }
  }
}

// ---------------- residual + LayerNorm (row of 128) ------------------------
__global__ __launch_bounds__(128) void ln_kernel(
    const float* __restrict__ base, const float* __restrict__ delta,
    const float* __restrict__ sc, const float* __restrict__ bi,
    float* __restrict__ seqF, half_t* __restrict__ seqH) {
  __shared__ float red[128];
  const int r = blockIdx.x, e = threadIdx.x;
  const size_t idx = (size_t)r * Ec + e;
  const float v = base[idx] + delta[idx];
  red[e] = v;
  __syncthreads();
  for (int s = 64; s > 0; s >>= 1) { if (e < s) red[e] += red[e + s]; __syncthreads(); }
  const float mean = red[0] * (1.f / 128.f);
  __syncthreads();
  const float dv = v - mean;
  red[e] = dv * dv;
  __syncthreads();
  for (int s = 64; s > 0; s >>= 1) { if (e < s) red[e] += red[e + s]; __syncthreads(); }
  const float var = red[0] * (1.f / 128.f);
  const float o = dv * rsqrtf(var + 1e-5f) * sc[e] + bi[e];
  seqF[idx] = o;
  seqH[idx] = (half_t)o;
}

// ---------------- split-K reduction of the final head ----------------------
__global__ __launch_bounds__(128) void reduce_final_kernel(
    const float* __restrict__ part, const float* __restrict__ ob,
    float* __restrict__ out) {
  const int i = blockIdx.x * 128 + threadIdx.x;   // 32*128 = 4096
  float s = ob[i & (OUTc - 1)];
  for (int z = 0; z < KSPLIT; ++z) s += part[(size_t)z * Bc * OUTc + i];
  out[i] = s;
}

// ---------------------------------------------------------------------------
extern "C" void kernel_launch(void* const* d_in, const int* in_sizes, int n_in,
                              void* d_out, int out_size, void* d_ws, size_t ws_size,
                              hipStream_t stream) {
  (void)in_sizes; (void)n_in; (void)out_size; (void)ws_size;
  const float* x          = (const float*)d_in[0];
  const float* prototypes = (const float*)d_in[1];
  const float* emb_w      = (const float*)d_in[2];
  const float* emb_b      = (const float*)d_in[3];
  const float* proj_w     = (const float*)d_in[4];
  const float* proj_b     = (const float*)d_in[5];
  const float* in_proj_w  = (const float*)d_in[6];
  const float* in_proj_b  = (const float*)d_in[7];
  const float* out_proj_w = (const float*)d_in[8];
  const float* out_proj_b = (const float*)d_in[9];
  const float* ln1_s      = (const float*)d_in[10];
  const float* ln1_b      = (const float*)d_in[11];
  const float* ffn_w1     = (const float*)d_in[12];
  const float* ffn_b1     = (const float*)d_in[13];
  const float* ffn_w2     = (const float*)d_in[14];
  const float* ffn_b2     = (const float*)d_in[15];
  const float* ln2_s      = (const float*)d_in[16];
  const float* ln2_b      = (const float*)d_in[17];
  const float* out_w      = (const float*)d_in[18];
  const float* out_b      = (const float*)d_in[19];

  // workspace carve (~60 MB total)
  char* ws = (char*)d_ws;
  size_t off = 0;
  auto carve = [&](size_t bytes) {
    char* p = ws + off;
    off += (bytes + 255) & ~(size_t)255;
    return (void*)p;
  };
  float*  pnorm = (float*)carve((size_t)Dc * Kc * Pc * 4);
  float*  seqF  = (float*)carve((size_t)Rows * Ec * 4);
  half_t* seqH  = (half_t*)carve((size_t)Rows * Ec * 2);
  half_t* qkvH  = (half_t*)carve((size_t)Rows * 3 * Ec * 2);
  half_t* ctxH  = (half_t*)carve((size_t)Rows * Ec * 2);
  float*  tmpF  = (float*)carve((size_t)Rows * Ec * 4);
  half_t* hH    = (half_t*)carve((size_t)Rows * Ic * 2);
  float*  partF = (float*)carve((size_t)KSPLIT * Bc * OUTc * 4);

  proto_norm_kernel<<<(Dc * Kc + 127) / 128, 128, 0, stream>>>(prototypes, pnorm);
  embed_kernel<<<Rows, 128, 0, stream>>>(x, pnorm, emb_w, emb_b, proj_w, proj_b,
                                         seqF, seqH);
  for (int l = 0; l < NLc; ++l) {
    // QKV: [24576,128] x [128,384]
    gemm_wmma_kernel<0><<<dim3(Rows / 16, 6, 1), 128, 0, stream>>>(
        seqH, in_proj_w + (size_t)l * 3 * Ec * Ec, in_proj_b + (size_t)l * 3 * Ec,
        nullptr, qkvH, Rows, 3 * Ec, Ec);
    // attention
    attn_kernel<<<Bc * Hc, 128, 0, stream>>>(qkvH, ctxH);
    // out-proj: [24576,128] x [128,128] -> tmpF
    gemm_wmma_kernel<1><<<dim3(Rows / 16, 2, 1), 128, 0, stream>>>(
        ctxH, out_proj_w + (size_t)l * Ec * Ec, out_proj_b + (size_t)l * Ec,
        tmpF, nullptr, Rows, Ec, Ec);
    ln_kernel<<<Rows, 128, 0, stream>>>(seqF, tmpF, ln1_s + (size_t)l * Ec,
                                        ln1_b + (size_t)l * Ec, seqF, seqH);
    // FFN1 + ELU: [24576,128] x [128,64] -> hH (f16)
    gemm_wmma_kernel<2><<<dim3(Rows / 16, 1, 1), 128, 0, stream>>>(
        seqH, ffn_w1 + (size_t)l * Ic * Ec, ffn_b1 + (size_t)l * Ic,
        nullptr, hH, Rows, Ic, Ec);
    // FFN2: [24576,64] x [64,128] -> tmpF
    gemm_wmma_kernel<1><<<dim3(Rows / 16, 2, 1), 128, 0, stream>>>(
        hH, ffn_w2 + (size_t)l * Ec * Ic, ffn_b2 + (size_t)l * Ec,
        tmpF, nullptr, Rows, Ec, Ic);
    ln_kernel<<<Rows, 128, 0, stream>>>(seqF, tmpF, ln2_s + (size_t)l * Ec,
                                        ln2_b + (size_t)l * Ec, seqF, seqH);
  }
  // final head, split-K 48 ways: [32,98304] x [98304,128]
  gemm_wmma_kernel<3><<<dim3(Bc / 16, 2, KSPLIT), 128, 0, stream>>>(
      seqH, out_w, nullptr, partF, nullptr, Bc, OUTc, KF);
  reduce_final_kernel<<<Bc * OUTc / 128, 128, 0, stream>>>(partF, out_b,
                                                           (float*)d_out);
}
